// ModifiedGCN_62062277427824
// MI455X (gfx1250) — matrix operand order repaired and checked by reading, not compiled, not run on hardware
//
#include <hip/hip_runtime.h>

#define NN 100000
#define EE 1600000
#define FIN 512
#define HID 128
#define NC 40
#define NCP 48

typedef __attribute__((ext_vector_type(16))) __bf16 v16bf;
typedef __attribute__((ext_vector_type(8)))  __bf16 v8bf;
typedef __attribute__((ext_vector_type(4)))  __bf16 v4bf;
typedef __attribute__((ext_vector_type(8)))  float  v8f;
typedef __attribute__((ext_vector_type(4)))  unsigned int u32x4;
typedef __attribute__((ext_vector_type(8)))  int i32x8;
typedef __attribute__((ext_vector_type(4)))  int i32x4;

// ---------------- Tensor Data Mover: 2-D tile load (bf16, dim0-contiguous) ----------------
// D# per CDNA5 ISA ch.8: group0 {count=1, lds_addr, global_addr, type=2},
// group1 {data_size=2B, tensor_dim0/1, tile_dim0/1, dim0_stride}. Groups 2/3 zero (2-D).
__device__ __forceinline__ void tdm_load_2d(unsigned lds_addr, const void* gaddr,
                                            unsigned td0, unsigned td1,
                                            unsigned stride0, unsigned tl0, unsigned tl1) {
    unsigned long long ga = (unsigned long long)gaddr;
    u32x4 g0;
    g0[0] = 1u;                                   // count=1 (valid user descriptor)
    g0[1] = lds_addr;                             // LDS byte address
    g0[2] = (unsigned)(ga & 0xffffffffu);         // global_addr[31:0]
    g0[3] = (unsigned)((ga >> 32) & 0x1ffffffu) | (2u << 30);  // addr[56:32] | type=2
    i32x8 g1;
    g1[0] = (int)(1u << 16);                      // data_size=1 -> 2 bytes
    g1[1] = (int)(td0 << 16);                     // tensor_dim0[15:0]
    g1[2] = (int)((td0 >> 16) & 0xffffu) | (int)(td1 << 16);   // dim0 hi | dim1 lo
    g1[3] = (int)((td1 >> 16) & 0xffffu) | (int)(tl0 << 16);   // dim1 hi | tile_dim0
    g1[4] = (int)(tl1 & 0xffffu);                 // tile_dim1 (tile_dim2 = 0)
    g1[5] = (int)stride0;                         // tensor_dim0_stride[31:0]
    g1[6] = 0;
    g1[7] = 0;
    i32x4 z4 = {0, 0, 0, 0};
    i32x8 z8 = {0, 0, 0, 0, 0, 0, 0, 0};
    __builtin_amdgcn_tensor_load_to_lds(g0, g1, z4, z4, z8, 0);
}

// ---------------- degree / normalization ----------------
__global__ void k_deg_init(float* deg) {
    int i = blockIdx.x * blockDim.x + threadIdx.x;
    if (i < NN) deg[i] = 1.0f;  // self-loop
}

__global__ void k_deg_edge(const int* __restrict__ ei, float* deg) {
    int e = blockIdx.x * blockDim.x + threadIdx.x;
    if (e < EE) atomicAdd(&deg[ei[EE + e]], 1.0f);  // in-degree from col
}

__global__ void k_dis(float* deg) {
    int i = blockIdx.x * blockDim.x + threadIdx.x;
    if (i < NN) deg[i] = rsqrtf(deg[i]);
}

// ---------------- weight conversion to bf16, column-major ----------------
__global__ void k_cvtW1(const float* __restrict__ W1, __bf16* __restrict__ W1bf) {
    int idx = blockIdx.x * blockDim.x + threadIdx.x;        // 512*128
    if (idx >= FIN * HID) return;
    int k = idx >> 7, c = idx & 127;                        // coalesced read
    W1bf[c * FIN + k] = (__bf16)W1[idx];
}

__global__ void k_cvtW2(const float* __restrict__ W2, __bf16* __restrict__ W2bf) {
    int idx = blockIdx.x * blockDim.x + threadIdx.x;        // 128*48
    if (idx >= HID * NCP) return;
    int k = idx / NCP, c = idx % NCP;
    W2bf[c * HID + k] = (__bf16)((c < NC) ? W2[k * NC + c] : 0.0f);
}

// ---------------- shared GEMM helpers ----------------
// Stage 128x32 f32 tile -> bf16 LDS (row-major stride 32). Batched float4 loads.
template<bool FULL, bool RELU, int LD>
__device__ __forceinline__ void stage_a(const float* __restrict__ src, __bf16* __restrict__ dst,
                                        int row0, int k0, int tid) {
    float4 v[4];
#pragma unroll
    for (int i = 0; i < 4; ++i) {
        int slot = tid + i * 256;                 // 0..1023
        int r = slot >> 3, c4 = slot & 7;
        float4 t = make_float4(0.f, 0.f, 0.f, 0.f);
        if (FULL || (row0 + r) < NN)
            t = *(const float4*)&src[(long long)(row0 + r) * LD + k0 + c4 * 4];
        if (RELU) {
            t.x = fmaxf(t.x, 0.f); t.y = fmaxf(t.y, 0.f);
            t.z = fmaxf(t.z, 0.f); t.w = fmaxf(t.w, 0.f);
        }
        v[i] = t;
    }
#pragma unroll
    for (int i = 0; i < 4; ++i) {
        int slot = tid + i * 256;
        int r = slot >> 3, c4 = slot & 7;
        v4bf h = { (__bf16)v[i].x, (__bf16)v[i].y, (__bf16)v[i].z, (__bf16)v[i].w };
        *(v4bf*)&dst[r * 32 + c4 * 4] = h;
    }
}

template<int NT>
__device__ __forceinline__ void mma_tiles(const __bf16* __restrict__ sA,
                                          const __bf16* __restrict__ sB,
                                          int arow, int kb, int koff, int lane, v8f* acc) {
    v8bf alo = *(const v8bf*)&sA[arow * 32 + koff];
    v8bf ahi = *(const v8bf*)&sA[arow * 32 + koff + 8];
    v16bf a = __builtin_shufflevector(alo, ahi, 0,1,2,3,4,5,6,7,8,9,10,11,12,13,14,15);
#pragma unroll
    for (int n = 0; n < NT; ++n) {
        v16bf b = *(const v16bf*)&sB[(n * 16 + (lane & 15)) * 32 + koff];
        acc[n] = __builtin_amdgcn_wmma_f32_16x16x32_bf16(
            false, a, false, b, (short)0, acc[n], false, false);
    }
}

// ---------------- GEMM1: H1 = X @ W1, 128x128 tile per block, TDM B, dbl-buffer ----------------
template<bool FULL>
__global__ void k_gemm1(const float* __restrict__ X, const __bf16* __restrict__ W1bf,
                        float* __restrict__ H1, int blk0) {
    __shared__ __align__(128) __bf16 sA[2][128 * 32];
    __shared__ __align__(128) __bf16 sB[2][128 * 32];
    const int tid = threadIdx.x, lane = tid & 31, wav = tid >> 5;
    const int row0 = (blk0 + blockIdx.x) * 128;

    v8f acc[8];
#pragma unroll
    for (int n = 0; n < 8; ++n) acc[n] = (v8f){0.f,0.f,0.f,0.f,0.f,0.f,0.f,0.f};

    const int arow = wav * 16 + (lane & 15);
    const int kb   = (lane >> 4) * 8;
    const int koff = (lane >> 4) * 16;

    if (wav == 0)
        tdm_load_2d((unsigned)(unsigned long long)&sB[0][0], W1bf, FIN, HID, FIN, 32, HID);
    stage_a<FULL, false, FIN>(X, sA[0], row0, 0, tid);
    if (wav == 0) __builtin_amdgcn_s_wait_tensorcnt(0);
    __syncthreads();

    for (int kc = 0; kc < FIN / 32; ++kc) {
        const int cur = kc & 1, nxt = cur ^ 1;
        if (kc + 1 < FIN / 32) {
            if (wav == 0)
                tdm_load_2d((unsigned)(unsigned long long)&sB[nxt][0],
                            W1bf + (kc + 1) * 32, FIN, HID, FIN, 32, HID);
            stage_a<FULL, false, FIN>(X, sA[nxt], row0, (kc + 1) * 32, tid);
        }
        mma_tiles<8>(sA[cur], sB[cur], arow, kb, koff, lane, acc);
        if (wav == 0 && kc + 1 < FIN / 32) __builtin_amdgcn_s_wait_tensorcnt(0);
        __syncthreads();
    }

    const int lM = (lane >> 4) * 8, lN = lane & 15;
#pragma unroll
    for (int n = 0; n < 8; ++n)
#pragma unroll
        for (int r = 0; r < 8; ++r) {
            int row = row0 + wav * 16 + r + lM;
            if (FULL || row < NN) H1[(long long)row * HID + n * 16 + lN] = acc[n][r];
        }
}

// ---------------- GEMM2: Z = relu(A1) @ W2 (N padded to 48) ----------------
template<bool FULL>
__global__ void k_gemm2(const float* __restrict__ A1, const __bf16* __restrict__ W2bf,
                        float* __restrict__ Z, int blk0) {
    __shared__ __align__(128) __bf16 sA[2][128 * 32];
    __shared__ __align__(128) __bf16 sB[2][NCP * 32];
    const int tid = threadIdx.x, lane = tid & 31, wav = tid >> 5;
    const int row0 = (blk0 + blockIdx.x) * 128;

    v8f acc[3];
#pragma unroll
    for (int n = 0; n < 3; ++n) acc[n] = (v8f){0.f,0.f,0.f,0.f,0.f,0.f,0.f,0.f};

    const int arow = wav * 16 + (lane & 15);
    const int kb   = (lane >> 4) * 8;
    const int koff = (lane >> 4) * 16;

    if (wav == 0)
        tdm_load_2d((unsigned)(unsigned long long)&sB[0][0], W2bf, HID, NCP, HID, 32, NCP);
    stage_a<FULL, true, HID>(A1, sA[0], row0, 0, tid);
    if (wav == 0) __builtin_amdgcn_s_wait_tensorcnt(0);
    __syncthreads();

    for (int kc = 0; kc < HID / 32; ++kc) {
        const int cur = kc & 1, nxt = cur ^ 1;
        if (kc + 1 < HID / 32) {
            if (wav == 0)
                tdm_load_2d((unsigned)(unsigned long long)&sB[nxt][0],
                            W2bf + (kc + 1) * 32, HID, NCP, HID, 32, NCP);
            stage_a<FULL, true, HID>(A1, sA[nxt], row0, (kc + 1) * 32, tid);
        }
        mma_tiles<3>(sA[cur], sB[cur], arow, kb, koff, lane, acc);
        if (wav == 0 && kc + 1 < HID / 32) __builtin_amdgcn_s_wait_tensorcnt(0);
        __syncthreads();
    }

    const int lM = (lane >> 4) * 8, lN = lane & 15;
#pragma unroll
    for (int n = 0; n < 3; ++n)
#pragma unroll
        for (int r = 0; r < 8; ++r) {
            int row = row0 + wav * 16 + r + lM;
            int col = n * 16 + lN;
            if ((FULL || row < NN) && col < NC)
                Z[(long long)row * NC + col] = acc[n][r];
        }
}

// ---------------- aggregation layer 1 ----------------
__global__ void k_init1(const float* __restrict__ H1, const float* __restrict__ dis,
                        const float* __restrict__ b1, float* __restrict__ A1) {
    long long idx = (long long)blockIdx.x * blockDim.x + threadIdx.x;   // N*32 float4s
    if (idx >= (long long)NN * 32) return;
    int node = (int)(idx >> 5);
    int q = (int)(idx & 31);
    float d = dis[node], dd = d * d;
    float4 h = ((const float4*)H1)[idx];
    float4 b = ((const float4*)b1)[q];
    float4 o = make_float4(dd * h.x + b.x, dd * h.y + b.y, dd * h.z + b.z, dd * h.w + b.w);
    ((float4*)A1)[idx] = o;
}

__global__ void k_edge1(const int* __restrict__ ei, const float* __restrict__ dis,
                        const float* __restrict__ H1, float* __restrict__ A1) {
    long long idx = (long long)blockIdx.x * blockDim.x + threadIdx.x;   // E*32
    if (idx >= (long long)EE * 32) return;
    int e = (int)(idx >> 5);
    int q = (int)(idx & 31);
    int r = ei[e], c = ei[EE + e];
    float nrm = dis[r] * dis[c];
    float4 h = ((const float4*)(H1 + (long long)r * HID))[q];
    float* o = A1 + (long long)c * HID + q * 4;
    atomicAdd(o + 0, nrm * h.x);
    atomicAdd(o + 1, nrm * h.y);
    atomicAdd(o + 2, nrm * h.z);
    atomicAdd(o + 3, nrm * h.w);
}

// ---------------- aggregation layer 2 (into d_out) ----------------
__global__ void k_init2(const float* __restrict__ Z, const float* __restrict__ dis,
                        const float* __restrict__ b2, float* __restrict__ out) {
    long long idx = (long long)blockIdx.x * blockDim.x + threadIdx.x;   // N*10 float4s
    if (idx >= (long long)NN * (NC / 4)) return;
    int node = (int)(idx / (NC / 4));
    int q = (int)(idx - (long long)node * (NC / 4));
    float d = dis[node], dd = d * d;
    float4 z = ((const float4*)Z)[idx];
    float4 b = ((const float4*)b2)[q];
    float4 o = make_float4(dd * z.x + b.x, dd * z.y + b.y, dd * z.z + b.z, dd * z.w + b.w);
    ((float4*)out)[idx] = o;
}

__global__ void k_edge2(const int* __restrict__ ei, const float* __restrict__ dis,
                        const float* __restrict__ Z, float* __restrict__ out) {
    long long idx = (long long)blockIdx.x * blockDim.x + threadIdx.x;   // E*40
    if (idx >= (long long)EE * NC) return;
    int e = (int)(idx / NC);
    int f = (int)(idx - (long long)e * NC);
    int r = ei[e], c = ei[EE + e];
    float nrm = dis[r] * dis[c];
    atomicAdd(out + (long long)c * NC + f, nrm * Z[(long long)r * NC + f]);
}

// ---------------- in-place log_softmax, one wave per node ----------------
__global__ void k_logsoftmax(float* __restrict__ out) {
    int node = blockIdx.x * 8 + (threadIdx.x >> 5);
    int lane = threadIdx.x & 31;
    if (node >= NN) return;
    float* p = out + (long long)node * NC;
    float v0 = p[lane];
    float v1 = (lane < NC - 32) ? p[32 + lane] : -__builtin_inff();
    float m = fmaxf(v0, v1);
#pragma unroll
    for (int off = 16; off; off >>= 1) m = fmaxf(m, __shfl_xor(m, off, 32));
    float s = __expf(v0 - m) + ((lane < NC - 32) ? __expf(v1 - m) : 0.0f);
#pragma unroll
    for (int off = 16; off; off >>= 1) s += __shfl_xor(s, off, 32);
    float lse = m + __logf(s);
    p[lane] = v0 - lse;
    if (lane < NC - 32) p[32 + lane] = v1 - lse;
}

extern "C" void kernel_launch(void* const* d_in, const int* in_sizes, int n_in,
                              void* d_out, int out_size, void* d_ws, size_t ws_size,
                              hipStream_t stream) {
    const float* X  = (const float*)d_in[0];
    const int*   EI = (const int*)d_in[1];      // edge_index [2, E]
    const float* W1 = (const float*)d_in[2];
    const float* B1 = (const float*)d_in[3];
    const float* W2 = (const float*)d_in[4];
    const float* B2 = (const float*)d_in[5];
    float* OUT = (float*)d_out;

    // workspace carve-up
    char* ws = (char*)d_ws;
    const size_t MPAD = ((NN + 127) / 128) * 128;            // 100096
    size_t off = 0;
    float* dis = (float*)(ws + off);
    off += ((size_t)NN * 4 + 511) & ~(size_t)511;
    __bf16* W1bf = (__bf16*)(ws + off);                       // 512x128 col-major
    off += (size_t)FIN * HID * 2;
    __bf16* W2bf = (__bf16*)(ws + off);                       // 128x48 col-major
    off += (size_t)HID * NCP * 2;
    off = (off + 511) & ~(size_t)511;
    float* H1 = (float*)(ws + off);                           // MPAD x 128
    off += MPAD * HID * 4;
    float* A1 = (float*)(ws + off);                           // MPAD x 128
    float* Z  = H1;                                           // alias: H1 dead after edge1

    const int T = 256;
    const int NFULL = NN / 128;                               // 781 full row-blocks
    const int NREM  = (int)(MPAD / 128) - NFULL;              // 1 partial block

    k_deg_init<<<(NN + T - 1) / T, T, 0, stream>>>(dis);
    k_deg_edge<<<(EE + T - 1) / T, T, 0, stream>>>(EI, dis);
    k_dis<<<(NN + T - 1) / T, T, 0, stream>>>(dis);

    k_cvtW1<<<(FIN * HID + T - 1) / T, T, 0, stream>>>(W1, W1bf);
    k_cvtW2<<<(HID * NCP + T - 1) / T, T, 0, stream>>>(W2, W2bf);

    k_gemm1<true><<<NFULL, T, 0, stream>>>(X, W1bf, H1, 0);
    k_gemm1<false><<<NREM, T, 0, stream>>>(X, W1bf, H1, NFULL);

    long long n1 = (long long)NN * 32;
    k_init1<<<(unsigned)((n1 + T - 1) / T), T, 0, stream>>>(H1, dis, B1, A1);
    long long ne1 = (long long)EE * 32;
    k_edge1<<<(unsigned)((ne1 + T - 1) / T), T, 0, stream>>>(EI, dis, H1, A1);

    k_gemm2<true><<<NFULL, T, 0, stream>>>(A1, W2bf, Z, 0);
    k_gemm2<false><<<NREM, T, 0, stream>>>(A1, W2bf, Z, NFULL);

    long long n2 = (long long)NN * (NC / 4);
    k_init2<<<(unsigned)((n2 + T - 1) / T), T, 0, stream>>>(Z, dis, B2, OUT);
    long long ne2 = (long long)EE * NC;
    k_edge2<<<(unsigned)((ne2 + T - 1) / T), T, 0, stream>>>(EI, dis, Z, OUT);

    k_logsoftmax<<<(NN + 7) / 8, T, 0, stream>>>(OUT);
}
